// Global_Critic_79224966742444
// MI455X (gfx1250) — compile-verified, matrix operand order checked by reading, not run on hardware
//
#include <hip/hip_runtime.h>
#include <hip/hip_bf16.h>

typedef _Float16 h16;
typedef __attribute__((ext_vector_type(16))) _Float16 v16h;
typedef __attribute__((ext_vector_type(8)))  _Float16 v8h;
typedef __attribute__((ext_vector_type(8)))  float    v8f;

#define NAG 8
#define BB  32768
#define OBSF 128
#define ACTF 32
#define EMB 128
#define NHEAD 4
#define DHEAD 32
#define LDSPAD 8   // halves of padding per LDS weight row (16B -> bank-spread)

// ---------------------------------------------------------------------------
// Async global->LDS copy (gfx1250): 16B per lane, ASYNCcnt-tracked, no VGPRs.
// ---------------------------------------------------------------------------
__device__ __forceinline__ void async_ld_b128(unsigned lds_off, const void* g) {
  asm volatile("global_load_async_to_lds_b128 %0, %1, off"
               :: "v"(lds_off), "v"(g) : "memory");
}
__device__ __forceinline__ void wait_async0() {
  asm volatile("s_wait_asynccnt 0x0" ::: "memory");
}

// ---------------------------------------------------------------------------
// Fragment loaders (wave32 WMMA 16x16x32 f16 layout):
// A (16x32): lane l<16 holds row M=l, K {k0+0..7, k0+16..23}; lane l>=16 holds
// row M=l-16, K {k0+8..15, k0+24..31}.  B fragment mirrors this with N=lane.
// ---------------------------------------------------------------------------
__device__ __forceinline__ v16h frag_from_f16(const h16* p, int hi) {
  v8h lo = *(const v8h*)(p + hi * 8);
  v8h hh = *(const v8h*)(p + 16 + hi * 8);
  return __builtin_shufflevector(lo, hh, 0,1,2,3,4,5,6,7,8,9,10,11,12,13,14,15);
}

__device__ __forceinline__ v16h frag_from_f32(const float* p, int hi) {
  float4 a = *(const float4*)(p + hi * 8);
  float4 b = *(const float4*)(p + hi * 8 + 4);
  float4 c = *(const float4*)(p + 16 + hi * 8);
  float4 d = *(const float4*)(p + 16 + hi * 8 + 4);
  v16h r;
  r[0]=(h16)a.x; r[1]=(h16)a.y; r[2]=(h16)a.z; r[3]=(h16)a.w;
  r[4]=(h16)b.x; r[5]=(h16)b.y; r[6]=(h16)b.z; r[7]=(h16)b.w;
  r[8]=(h16)c.x; r[9]=(h16)c.y; r[10]=(h16)c.z; r[11]=(h16)c.w;
  r[12]=(h16)d.x; r[13]=(h16)d.y; r[14]=(h16)d.z; r[15]=(h16)d.w;
  return r;
}

// ---------------------------------------------------------------------------
// BatchNorm statistics: one block per (agent, feature), reduce over B.
// ---------------------------------------------------------------------------
__global__ void k_bnstats(const float* __restrict__ obs, const float* __restrict__ act,
                          float* __restrict__ mean, float* __restrict__ rstd) {
  const int f = blockIdx.x % 160;
  const int n = blockIdx.x / 160;
  const float* src; int ld;
  if (f < OBSF) { src = obs + (long)n * BB * OBSF + f;          ld = OBSF; }
  else          { src = act + (long)n * BB * ACTF + (f - OBSF); ld = ACTF; }
  float s = 0.f, sq = 0.f;
  for (int b = threadIdx.x; b < BB; b += 256) {
    float x = src[(long)b * ld];
    s += x; sq += x * x;
  }
  __shared__ float sh[256], sh2[256];
  sh[threadIdx.x] = s; sh2[threadIdx.x] = sq;
  __syncthreads();
  for (int off = 128; off > 0; off >>= 1) {
    if (threadIdx.x < off) {
      sh[threadIdx.x]  += sh[threadIdx.x + off];
      sh2[threadIdx.x] += sh2[threadIdx.x + off];
    }
    __syncthreads();
  }
  if (threadIdx.x == 0) {
    float mu  = sh[0] * (1.f / (float)BB);
    float var = sh2[0] * (1.f / (float)BB) - mu * mu;
    mean[n * 160 + f] = mu;
    rstd[n * 160 + f] = rsqrtf(var + 1e-5f);
  }
}

// ---------------------------------------------------------------------------
// Fold BN affine+stats into weights (f16 weights + adjusted f32 bias).
// ---------------------------------------------------------------------------
__global__ void k_fold(const float* __restrict__ Wsrc, const float* __restrict__ gamma,
                       const float* __restrict__ beta, const float* __restrict__ bsrc,
                       const float* __restrict__ mean, const float* __restrict__ rstd,
                       int K, h16* __restrict__ Wdst, float* __restrict__ bdst) {
  const int e = blockIdx.x % EMB;
  const int n = blockIdx.x / EMB;
  const int f = threadIdx.x;
  float part = 0.f;
  if (f < K) {
    float w  = Wsrc[((long)n * EMB + e) * K + f];
    float g  = gamma[n * K + f];
    float bt = beta[n * K + f];
    float mu = mean[n * 160 + f];
    float rs = rstd[n * 160 + f];
    Wdst[((long)n * EMB + e) * K + f] = (h16)(w * g * rs);
    part = w * (bt - mu * rs * g);
  }
  __shared__ float sh[256];
  sh[threadIdx.x] = part;
  __syncthreads();
  for (int off = 128; off > 0; off >>= 1) {
    if (threadIdx.x < off) sh[threadIdx.x] += sh[threadIdx.x + off];
    __syncthreads();
  }
  if (threadIdx.x == 0) bdst[n * EMB + e] = bsrc[n * EMB + e] + sh[0];
}

__global__ void k_cvt(const float* __restrict__ src, h16* __restrict__ dst, long n) {
  long i = (long)blockIdx.x * blockDim.x + threadIdx.x;
  if (i < n) dst[i] = (h16)src[i];
}

// ---------------------------------------------------------------------------
// WMMA GEMM, f32 activations converted on the fly.  The block stages its
// 64-column weight tile in LDS via async global->LDS, then each of the 8
// waves computes a 32x64 tile (2 A-frags x 4 B-frags = 8 WMMA per k-tile).
// grid: (B/256, outTiles(=2), N); block: 256.
// ---------------------------------------------------------------------------
__global__ void __launch_bounds__(256)
k_gemm_f32a(const float* __restrict__ S0, int lds0, long s0Stride,
            const float* __restrict__ S1, int lds1, long s1Stride, int kt0,
            int ktiles,
            const h16* __restrict__ W, int ldw, long wStride,
            const float* __restrict__ bias, long bStride,
            h16* __restrict__ C, int ldc, int cCol0, long cStride, float leak) {
  __shared__ __align__(16) h16 Wlds[64 * (256 + LDSPAD)];
  const int lane = threadIdx.x & 31;
  const int wave = threadIdx.x >> 5;
  const int n    = blockIdx.z;
  const int row0 = blockIdx.x * 256 + wave * 32;
  const int col0 = blockIdx.y * 64;
  const int m  = lane & 15;
  const int hi = lane >> 4;
  const int Kfull = ktiles * 32;
  const int LP = Kfull + LDSPAD;

  // --- stage W[col0:col0+64, 0:K] into LDS (async, 16B chunks) ---
  {
    const h16* Wb = W + (long)n * wStride;
    const unsigned base = (unsigned)(uintptr_t)Wlds;
    const int chunksPerRow = Kfull >> 3;            // 16B chunks per row
    const int total = 64 * chunksPerRow;
    for (int c = threadIdx.x; c < total; c += 256) {
      int col = c / chunksPerRow;
      int ko  = (c - col * chunksPerRow) << 3;
      async_ld_b128(base + (unsigned)(col * LP + ko) * 2,
                    Wb + (long)(col0 + col) * ldw + ko);
    }
    wait_async0();
    __syncthreads();
  }

  v8f acc[2][4] = {{{}, {}, {}, {}}, {{}, {}, {}, {}}};
  for (int kt = 0; kt < ktiles; ++kt) {
    const int k0 = kt * 32;
    const float *ap0, *ap1;
    if (kt < kt0) {
      ap0 = S0 + (long)n * s0Stride + (long)(row0 + m) * lds0 + k0;
      ap1 = S0 + (long)n * s0Stride + (long)(row0 + 16 + m) * lds0 + k0;
    } else {
      ap0 = S1 + (long)n * s1Stride + (long)(row0 + m) * lds1 + (k0 - kt0 * 32);
      ap1 = S1 + (long)n * s1Stride + (long)(row0 + 16 + m) * lds1 + (k0 - kt0 * 32);
    }
    v16h a0 = frag_from_f32(ap0, hi);
    v16h a1 = frag_from_f32(ap1, hi);
#pragma unroll
    for (int j = 0; j < 4; ++j) {
      const int wr = (j * 16 + m) * LP + k0 + hi * 8;
      v8h lo = *(const v8h*)&Wlds[wr];
      v8h hh = *(const v8h*)&Wlds[wr + 16];
      v16h b = __builtin_shufflevector(lo, hh, 0,1,2,3,4,5,6,7,8,9,10,11,12,13,14,15);
      acc[0][j] = __builtin_amdgcn_wmma_f32_16x16x32_f16(false, a0, false, b,
                                                         (short)0, acc[0][j], false, false);
      acc[1][j] = __builtin_amdgcn_wmma_f32_16x16x32_f16(false, a1, false, b,
                                                         (short)0, acc[1][j], false, false);
    }
  }

  const float* bb = bias ? bias + (long)n * bStride : nullptr;
  h16* Cb = C + (long)n * cStride;
#pragma unroll
  for (int s = 0; s < 2; ++s) {
#pragma unroll
    for (int j = 0; j < 4; ++j) {
      const int cc = col0 + j * 16 + m;
      const float bv = bb ? bb[cc] : 0.f;
#pragma unroll
      for (int r = 0; r < 8; ++r) {
        float x = acc[s][j][r] + bv;
        x = (x > 0.f) ? x : x * leak;
        Cb[(long)(row0 + s * 16 + hi * 8 + r) * ldc + (cCol0 + cc)] = (h16)x;
      }
    }
  }
}

// ---------------------------------------------------------------------------
// WMMA GEMM, f16 activations; same LDS weight staging + 32x64 wave tile.
// ---------------------------------------------------------------------------
__global__ void __launch_bounds__(256)
k_gemm_f16(const h16* __restrict__ A, int lda, long aStride,
           const h16* __restrict__ W, int ldw, long wStride,
           const float* __restrict__ bias, long bStride,
           h16* __restrict__ C, int ldc, int cCol0, long cStride,
           int ktiles, float leak) {
  __shared__ __align__(16) h16 Wlds[64 * (256 + LDSPAD)];
  const int lane = threadIdx.x & 31;
  const int wave = threadIdx.x >> 5;
  const int n    = blockIdx.z;
  const int row0 = blockIdx.x * 256 + wave * 32;
  const int col0 = blockIdx.y * 64;
  const int m  = lane & 15;
  const int hi = lane >> 4;
  const int Kfull = ktiles * 32;
  const int LP = Kfull + LDSPAD;

  {
    const h16* Wb = W + (long)n * wStride;
    const unsigned base = (unsigned)(uintptr_t)Wlds;
    const int chunksPerRow = Kfull >> 3;
    const int total = 64 * chunksPerRow;
    for (int c = threadIdx.x; c < total; c += 256) {
      int col = c / chunksPerRow;
      int ko  = (c - col * chunksPerRow) << 3;
      async_ld_b128(base + (unsigned)(col * LP + ko) * 2,
                    Wb + (long)(col0 + col) * ldw + ko);
    }
    wait_async0();
    __syncthreads();
  }

  const h16* Ab0 = A + (long)n * aStride + (long)(row0 + m) * lda;
  const h16* Ab1 = A + (long)n * aStride + (long)(row0 + 16 + m) * lda;
  v8f acc[2][4] = {{{}, {}, {}, {}}, {{}, {}, {}, {}}};

  for (int kt = 0; kt < ktiles; ++kt) {
    const int k0 = kt * 32;
    v16h a0 = frag_from_f16(Ab0 + k0, hi);
    v16h a1 = frag_from_f16(Ab1 + k0, hi);
#pragma unroll
    for (int j = 0; j < 4; ++j) {
      const int wr = (j * 16 + m) * LP + k0 + hi * 8;
      v8h lo = *(const v8h*)&Wlds[wr];
      v8h hh = *(const v8h*)&Wlds[wr + 16];
      v16h b = __builtin_shufflevector(lo, hh, 0,1,2,3,4,5,6,7,8,9,10,11,12,13,14,15);
      acc[0][j] = __builtin_amdgcn_wmma_f32_16x16x32_f16(false, a0, false, b,
                                                         (short)0, acc[0][j], false, false);
      acc[1][j] = __builtin_amdgcn_wmma_f32_16x16x32_f16(false, a1, false, b,
                                                         (short)0, acc[1][j], false, false);
    }
  }

  const float* bb = bias ? bias + (long)n * bStride : nullptr;
  h16* Cb = C + (long)n * cStride;
#pragma unroll
  for (int s = 0; s < 2; ++s) {
#pragma unroll
    for (int j = 0; j < 4; ++j) {
      const int cc = col0 + j * 16 + m;
      const float bv = bb ? bb[cc] : 0.f;
#pragma unroll
      for (int r = 0; r < 8; ++r) {
        float x = acc[s][j][r] + bv;
        x = (x > 0.f) ? x : x * leak;
        Cb[(long)(row0 + s * 16 + hi * 8 + r) * ldc + (cCol0 + cc)] = (h16)x;
      }
    }
  }
}

// ---------------------------------------------------------------------------
// Leave-one-out attention across N=8 agents.  One wave per batch element b.
// ---------------------------------------------------------------------------
__global__ void __launch_bounds__(256)
k_attn(const h16* __restrict__ q16, const h16* __restrict__ k16,
       const h16* __restrict__ v16, h16* __restrict__ fin) {
  __shared__ __align__(16) h16 kv[8][2][NAG][EMB];  // 32 KB
  const int lane = threadIdx.x & 31;
  const int wave = threadIdx.x >> 5;
  const long b = (long)blockIdx.x * 8 + wave;

#pragma unroll
  for (int i = 0; i < 4; ++i) {
    int chunk = lane + 32 * i;          // 0..127 (8 agents x 16 chunks)
    int mAg = chunk >> 4;
    int part = chunk & 15;
    const uint4* gk = (const uint4*)(k16 + ((long)mAg * BB + b) * EMB + part * 8);
    const uint4* gv = (const uint4*)(v16 + ((long)mAg * BB + b) * EMB + part * 8);
    *(uint4*)&kv[wave][0][mAg][part * 8] = *gk;
    *(uint4*)&kv[wave][1][mAg][part * 8] = *gv;
  }
  __syncthreads();

  const int n = lane >> 2;
  const int h = lane & 3;
  float q[DHEAD];
  const h16* qp = q16 + ((long)n * BB + b) * EMB + h * DHEAD;
#pragma unroll
  for (int d = 0; d < DHEAD; ++d) q[d] = (float)qp[d];

  float lg[NAG];
#pragma unroll
  for (int mAg = 0; mAg < NAG; ++mAg) {
    const h16* kp = &kv[wave][0][mAg][h * DHEAD];
    float acc = 0.f;
#pragma unroll
    for (int d = 0; d < DHEAD; ++d) acc += q[d] * (float)kp[d];
    lg[mAg] = acc * 0.17677669529663687f;  // 1/sqrt(32)
  }
  lg[n] = -__builtin_inff();
  float mx = -__builtin_inff();
#pragma unroll
  for (int mAg = 0; mAg < NAG; ++mAg) mx = fmaxf(mx, lg[mAg]);
  float ssum = 0.f;
#pragma unroll
  for (int mAg = 0; mAg < NAG; ++mAg) { lg[mAg] = __expf(lg[mAg] - mx); ssum += lg[mAg]; }
  float inv = 1.f / ssum;

  float xi[DHEAD];
#pragma unroll
  for (int d = 0; d < DHEAD; ++d) xi[d] = 0.f;
#pragma unroll
  for (int mAg = 0; mAg < NAG; ++mAg) {
    float a = lg[mAg] * inv;
    const h16* vp = &kv[wave][1][mAg][h * DHEAD];
#pragma unroll
    for (int d = 0; d < DHEAD; ++d) xi[d] += a * (float)vp[d];
  }
  h16* op = fin + ((long)n * BB + b) * 256 + h * DHEAD;
#pragma unroll
  for (int d = 0; d < DHEAD; ++d) op[d] = (h16)xi[d];
}

// ---------------------------------------------------------------------------
// Final: argmax over act (first-max), then the single selected f2 column.
// ---------------------------------------------------------------------------
__global__ void k_final(const float* __restrict__ act, const h16* __restrict__ h1,
                        const float* __restrict__ f2w, const float* __restrict__ f2b,
                        float* __restrict__ out) {
  long i = (long)blockIdx.x * blockDim.x + threadIdx.x;  // n*B + b
  if (i >= (long)NAG * BB) return;
  int n = (int)(i / BB);
  const float* ap = act + i * ACTF;
  int best = 0; float bv = ap[0];
#pragma unroll
  for (int a = 1; a < ACTF; ++a) { float x = ap[a]; if (x > bv) { bv = x; best = a; } }
  const h16* hp = h1 + i * EMB;
  const float* wp = f2w + ((long)n * ACTF + best) * EMB;
  float acc = 0.f;
#pragma unroll 8
  for (int e = 0; e < EMB; ++e) acc += (float)hp[e] * wp[e];
  out[i] = acc + f2b[n * ACTF + best];
}

// ---------------------------------------------------------------------------
extern "C" void kernel_launch(void* const* d_in, const int* in_sizes, int n_in,
                              void* d_out, int out_size, void* d_ws, size_t ws_size,
                              hipStream_t stream) {
  const float* obs   = (const float*)d_in[0];
  const float* act   = (const float*)d_in[1];
  const float* gGam  = (const float*)d_in[2];
  const float* gBet  = (const float*)d_in[3];
  const float* sGam  = (const float*)d_in[4];
  const float* sBet  = (const float*)d_in[5];
  const float* g_w   = (const float*)d_in[6];
  const float* g_b   = (const float*)d_in[7];
  const float* s_w   = (const float*)d_in[8];
  const float* s_b   = (const float*)d_in[9];
  const float* wq    = (const float*)d_in[10];
  const float* wk    = (const float*)d_in[11];
  const float* wv    = (const float*)d_in[12];
  const float* f1_w  = (const float*)d_in[13];
  const float* f1_b  = (const float*)d_in[14];
  const float* f2_w  = (const float*)d_in[15];
  const float* f2_b  = (const float*)d_in[16];
  float* out = (float*)d_out;

  char* p = (char*)d_ws;
  auto alloc = [&](size_t bytes) -> void* {
    void* r = (void*)p;
    p += (bytes + 255) & ~(size_t)255;
    return r;
  };
  float* mean  = (float*)alloc((size_t)NAG * 160 * 4);
  float* rstd  = (float*)alloc((size_t)NAG * 160 * 4);
  h16* g_w16   = (h16*)alloc((size_t)NAG * EMB * 160 * 2);
  float* g_b2  = (float*)alloc((size_t)NAG * EMB * 4);
  h16* s_w16   = (h16*)alloc((size_t)NAG * EMB * 128 * 2);
  float* s_b2  = (float*)alloc((size_t)NAG * EMB * 4);
  h16* wq16    = (h16*)alloc((size_t)EMB * EMB * 2);
  h16* wk16    = (h16*)alloc((size_t)EMB * EMB * 2);
  h16* wv16    = (h16*)alloc((size_t)EMB * EMB * 2);
  h16* f1_w16  = (h16*)alloc((size_t)NAG * EMB * 256 * 2);
  h16* e16     = (h16*)alloc((size_t)NAG * BB * EMB * 2);   // later reused as h1
  h16* q16     = (h16*)alloc((size_t)NAG * BB * EMB * 2);
  h16* k16     = (h16*)alloc((size_t)NAG * BB * EMB * 2);
  h16* v16     = (h16*)alloc((size_t)NAG * BB * EMB * 2);
  h16* fin16   = (h16*)alloc((size_t)NAG * BB * 256 * 2);   // [xi | s]
  h16* h1_16   = e16;                                        // alias (e dead after qkv)

  // 1) BN stats over B per (agent, feature)
  k_bnstats<<<NAG * 160, 256, 0, stream>>>(obs, act, mean, rstd);

  // 2) fold BN into g/s weights; convert qkv/f1 weights to f16
  k_fold<<<NAG * EMB, 256, 0, stream>>>(g_w, gGam, gBet, g_b, mean, rstd, 160, g_w16, g_b2);
  k_fold<<<NAG * EMB, 256, 0, stream>>>(s_w, sGam, sBet, s_b, mean, rstd, 128, s_w16, s_b2);
  k_cvt<<<64, 256, 0, stream>>>(wq, wq16, (long)EMB * EMB);
  k_cvt<<<64, 256, 0, stream>>>(wk, wk16, (long)EMB * EMB);
  k_cvt<<<64, 256, 0, stream>>>(wv, wv16, (long)EMB * EMB);
  k_cvt<<<1024, 256, 0, stream>>>(f1_w, f1_w16, (long)NAG * EMB * 256);

  dim3 gg(BB / 256, 2, NAG), bl(256);

  // 3) e = lrelu(bn(ga) @ g_w') ; s = lrelu(bn(obs) @ s_w') -> fin[:,128:256]
  k_gemm_f32a<<<gg, bl, 0, stream>>>(obs, OBSF, (long)BB * OBSF,
                                     act, ACTF, (long)BB * ACTF, 4, 5,
                                     g_w16, 160, (long)EMB * 160,
                                     g_b2, EMB,
                                     e16, EMB, 0, (long)BB * EMB, 0.01f);
  k_gemm_f32a<<<gg, bl, 0, stream>>>(obs, OBSF, (long)BB * OBSF,
                                     obs, OBSF, (long)BB * OBSF, 4, 4,
                                     s_w16, 128, (long)EMB * 128,
                                     s_b2, EMB,
                                     fin16, 256, 128, (long)BB * 256, 0.01f);

  // 4) q/k/v projections (weights shared across agents; lrelu only on v)
  k_gemm_f16<<<gg, bl, 0, stream>>>(e16, EMB, (long)BB * EMB, wq16, EMB, 0,
                                    nullptr, 0, q16, EMB, 0, (long)BB * EMB, 4, 1.0f);
  k_gemm_f16<<<gg, bl, 0, stream>>>(e16, EMB, (long)BB * EMB, wk16, EMB, 0,
                                    nullptr, 0, k16, EMB, 0, (long)BB * EMB, 4, 1.0f);
  k_gemm_f16<<<gg, bl, 0, stream>>>(e16, EMB, (long)BB * EMB, wv16, EMB, 0,
                                    nullptr, 0, v16, EMB, 0, (long)BB * EMB, 4, 0.01f);

  // 5) leave-one-out attention -> xi into fin[:,0:128]
  k_attn<<<BB / 8, 256, 0, stream>>>(q16, k16, v16, fin16);

  // 6) h1 = lrelu(fin @ f1_w + f1_b)  (h1 aliases the dead e16 buffer)
  k_gemm_f16<<<gg, bl, 0, stream>>>(fin16, 256, (long)BB * 256,
                                    f1_w16, 256, (long)EMB * 256,
                                    f1_b, EMB,
                                    h1_16, EMB, 0, (long)BB * EMB, 8, 0.01f);

  // 7) argmax(act) gather fused with the single needed f2 column
  k_final<<<(NAG * BB) / 256, 256, 0, stream>>>(act, h1_16, f2_w, f2_b, out);
}
